// TESTAM_91096256348184
// MI455X (gfx1250) — matrix operand order repaired and checked by reading, not compiled, not run on hardware
//
#include <hip/hip_runtime.h>

// ---------------------------------------------------------------------------
// TESTAM routing-score kernel for MI455X (gfx1250, wave32, WMMA bf16 + TDM)
// ---------------------------------------------------------------------------

typedef __attribute__((ext_vector_type(16))) __bf16 v16bf;
typedef __attribute__((ext_vector_type(8)))  __bf16 v8bf;
typedef __attribute__((ext_vector_type(4)))  __bf16 v4bf;
typedef __attribute__((ext_vector_type(8)))  float  v8f;
typedef __attribute__((ext_vector_type(4)))  float  v4f;
typedef __attribute__((ext_vector_type(4)))  unsigned int u32x4;
typedef __attribute__((ext_vector_type(8)))  int    i32x8;
typedef __attribute__((ext_vector_type(4)))  int    i32x4;

#if defined(__has_builtin)
#if __has_builtin(__builtin_amdgcn_tensor_load_to_lds) && __has_builtin(__builtin_amdgcn_s_wait_tensorcnt)
#define HAVE_TDM 1
#endif
#endif

#define B_    16
#define N_    1024
#define T_    48
#define HID_  64
#define MH_   32
#define MS_   20
#define NEXP_ 3
#define EPS_  1e-8f

#define NTHREADS 128   // 4 waves of 32

// LDS leading dimensions (elements). Byte strides must be multiples of 16 for b128.
#define HLD  72   // h / score tile (bf16) [48][64 used]      -> 144 B rows
#define QLD  40   // q,k tiles (bf16) [48][32]                ->  80 B rows
#define WTLD 72   // transposed W / vT tiles (bf16) [32][64]  -> 144 B rows
#define ELD  52   // energy / att scratch (f32) [48][<=48]    -> 208 B rows
#define MLD  33   // memory-side f32 tiles

// ---------------------------------------------------------------------------
// WMMA fragment helpers (ISA 7.12.2 layouts, wave32), vectorized b128 LDS I/O
// ---------------------------------------------------------------------------

// A fragment 16x32 bf16 from row-major buf[row][k]. Per lane: two contiguous
// 8-element runs: elems 0..7 = K(half*8..+7), elems 8..15 = K(16+half*8..+7).
__device__ __forceinline__ v16bf frag_a(const __bf16* buf, int ld, int row0, int k0, int lane) {
  const int hx = (lane >> 4) & 1;
  const int m  = lane & 15;
  const __bf16* r = buf + (row0 + m) * ld + k0 + hx * 8;
  v8bf lo = *(const v8bf*)(r);
  v8bf hi = *(const v8bf*)(r + 16);
  return __builtin_shufflevector(lo, hi, 0, 1, 2, 3, 4, 5, 6, 7, 8, 9, 10, 11, 12, 13, 14, 15);
}

// B fragment 32x16 where logical B[k][n] = buf[n][k] (buf row-major [N][K]).
// Per lane: two contiguous 8-element runs: elems 0..7 = K(half*8..+7), elems 8..15 = K(16+half*8..+7).
__device__ __forceinline__ v16bf frag_bT(const __bf16* buf, int ld, int k0, int col0, int lane) {
  const int hx = (lane >> 4) & 1;
  const int n  = lane & 15;
  const __bf16* r = buf + (col0 + n) * ld + k0 + hx * 8;
  v8bf lo = *(const v8bf*)(r);
  v8bf hi = *(const v8bf*)(r + 16);
  return __builtin_shufflevector(lo, hi, 0, 1, 2, 3, 4, 5, 6, 7, 8, 9, 10, 11, 12, 13, 14, 15);
}

// C/D 16x16 f32: VGPR r holds row (row0 + 8*half + r), col = lane&15.
__device__ __forceinline__ void store_d_f32(float* buf, int ld, int row0, int col0, int lane, v8f d) {
  const int hx = (lane >> 4) & 1;
  const int n  = lane & 15;
#pragma unroll
  for (int r = 0; r < 8; ++r)
    buf[(row0 + hx * 8 + r) * ld + col0 + n] = d[r];
}

// Row-major bf16 D store (strided b16, used for q/k).
__device__ __forceinline__ void store_d_bf16_rm(__bf16* buf, int ld, int row0, int col0, int lane, v8f d) {
  const int hx = (lane >> 4) & 1;
  const int n  = lane & 15;
#pragma unroll
  for (int r = 0; r < 8; ++r)
    buf[(row0 + hx * 8 + r) * ld + col0 + n] = (__bf16)d[r];
}

// Transposed bf16 D store: buf[col][row]; a lane's 8 rows are contiguous -> one b128.
__device__ __forceinline__ void store_d_bf16_tr(__bf16* buf, int ld, int row0, int col0, int lane, v8f d) {
  const int hx = (lane >> 4) & 1;
  const int n  = lane & 15;
  v8bf y;
#pragma unroll
  for (int r = 0; r < 8; ++r) y[r] = (__bf16)d[r];
  *(v8bf*)(buf + (col0 + n) * ld + row0 + hx * 8) = y;
}

__device__ __forceinline__ v8f wmma_bf16(v16bf a, v16bf b, v8f c) {
  return __builtin_amdgcn_wmma_f32_16x16x32_bf16(false, a, false, b, (short)0, c, false, false);
}

// ---------------------------------------------------------------------------
// TDM: async-load one 48x64 f32 tile (row-major, stride 64) into LDS staging.
// D# per ISA ch.8: group0 = {count/flags, lds_addr, global_addr, type=2},
// group1 = {data_size=4B, tensor_dim0=64, tensor_dim1=48, tile 64x48, stride0=64}.
// Toolchain builtin is the 6-arg form: (g0, g1, g2, g3, g4, cpol).
// ---------------------------------------------------------------------------
__device__ __forceinline__ void tdm_load_h_tile(const float* gsrc, float* lds_dst) {
#ifdef HAVE_TDM
  const unsigned long long ga = (unsigned long long)gsrc;
  const unsigned lds = (unsigned)(unsigned long long)lds_dst;   // low 32 bits = LDS offset
  u32x4 g0;
  g0[0] = 1u;                                                   // count=1, user mode
  g0[1] = lds;                                                  // lds_addr
  g0[2] = (unsigned)ga;                                         // global_addr[31:0]
  g0[3] = (unsigned)((ga >> 32) & 0x1FFFFFFull) | (2u << 30);   // addr[56:32] | type=2
  i32x8 g1;
  g1[0] = (int)(2u << 16);               // data_size=4B; no multicast/barrier/iterate/pad
  g1[1] = (int)((unsigned)HID_ << 16);   // tensor_dim0 = 64 (bits 63:48)
  g1[2] = (int)((unsigned)T_ << 16);     // tensor_dim1 = 48 (bits 111:80, low half)
  g1[3] = (int)((unsigned)HID_ << 16);   // tile_dim0 = 64 (bits 127:112)
  g1[4] = T_;                            // tile_dim1 = 48
  g1[5] = HID_;                          // tensor_dim0_stride = 64 (low 32)
  g1[6] = 0;
  g1[7] = 0;
  const i32x4 gz4 = {0, 0, 0, 0};
  const i32x8 gz8 = {0, 0, 0, 0, 0, 0, 0, 0};
  __builtin_amdgcn_tensor_load_to_lds(g0, g1, gz4, gz4, gz8, 0);
#else
  (void)gsrc; (void)lds_dst;
#endif
}

// ---------------------------------------------------------------------------
// Fused kernel: one block per (b,n); 4 waves; 3 experts processed in sequence.
// ---------------------------------------------------------------------------
__global__ __launch_bounds__(NTHREADS)
void testam_fused(const float* __restrict__ input,
                  const float* __restrict__ h0,
                  const float* __restrict__ h1,
                  const float* __restrict__ h2,
                  const float* __restrict__ memory,
                  const float* __restrict__ iq,
                  const float* __restrict__ hq,
                  const float* __restrict__ kw,
                  const float* __restrict__ vw,
                  float* __restrict__ out) {
  __shared__ float  s_bank[MS_ * MLD];                 // memory bank [20][32]
  __shared__ float  s_iq[2 * MH_];                     // input_query [2][32]
  __shared__ float  s_memb[T_ * MLD];                  // memories [48][32]
  __shared__ float  s_mnorm[T_];                       // |memories| per row
  __shared__ __align__(16) float  s_stage[T_ * HID_];  // TDM staging: h tile f32 [48][64]
  __shared__ __align__(16) __bf16 s_hs[T_ * HLD];      // union: h tile -> score tile (row-major)
  __shared__ __align__(16) __bf16 s_wT[NEXP_][MH_ * WTLD]; // Wq^T, Wk^T, Wv^T [32][64]
  __shared__ __align__(16) __bf16 s_q[T_ * QLD];       // q row-major [48][32]
  __shared__ __align__(16) __bf16 s_k[T_ * QLD];       // k row-major [48][32]
  __shared__ __align__(16) __bf16 s_vT[MH_ * WTLD];    // v^T [32][64] (cols 48..63 zero)
  __shared__ __align__(16) float  s_scr[T_ * ELD];     // union: energy [48][48] -> att [48][32]

  const int tid  = threadIdx.x;
  const int wave = tid >> 5;
  const int lane = tid & 31;
  const int bn   = blockIdx.x;                          // b*N + n
  const size_t hoff = (size_t)bn * T_ * HID_;

#ifdef HAVE_TDM
  // Prefetch expert-0 h tile under the memory-routing stage.
  if (wave == 0) tdm_load_h_tile(h0 + hoff, s_stage);
#endif

  // ---------------- Stage 0: memory routing (whole block) ----------------
  for (int i = tid; i < MS_ * MH_; i += NTHREADS)
    s_bank[(i / MH_) * MLD + (i % MH_)] = memory[i];
  for (int i = tid; i < 2 * MH_; i += NTHREADS)
    s_iq[i] = iq[i];
  __syncthreads();

  for (int t = tid; t < T_; t += NTHREADS) {
    const float* inp = input + ((size_t)bn * T_ + t) * 2;
    const float x0 = inp[0], x1 = inp[1];
    float qin[MH_];
#pragma unroll
    for (int m = 0; m < MH_; ++m)
      qin[m] = x0 * s_iq[m] + x1 * s_iq[MH_ + m];
    float en[MS_];
    float mx = -3.4e38f;
#pragma unroll
    for (int s = 0; s < MS_; ++s) {
      float acc = 0.f;
#pragma unroll
      for (int m = 0; m < MH_; ++m) acc += qin[m] * s_bank[s * MLD + m];
      en[s] = acc;
      mx = fmaxf(mx, acc);
    }
    float sum = 0.f;
#pragma unroll
    for (int s = 0; s < MS_; ++s) { en[s] = __expf(en[s] - mx); sum += en[s]; }
    const float inv = 1.f / sum;
    float nrm2 = 0.f;
#pragma unroll
    for (int m = 0; m < MH_; ++m) {
      float acc = 0.f;
#pragma unroll
      for (int s = 0; s < MS_; ++s) acc += en[s] * s_bank[s * MLD + m];
      acc *= inv;
      s_memb[t * MLD + m] = acc;
      nrm2 += acc * acc;
    }
    s_mnorm[t] = sqrtf(nrm2);
  }

  // ---------------- Expert loop ----------------
  for (int e = 0; e < NEXP_; ++e) {
    // ---- Stage A: h tile (via TDM staging) + weights -> bf16 LDS tiles ----
#ifdef HAVE_TDM
    if (wave == 0) __builtin_amdgcn_s_wait_tensorcnt(0);
    __syncthreads();   // TDM data visible; prior-expert consumers finished
    for (int i = tid * 4; i < T_ * HID_; i += NTHREADS * 4) {
      const v4f x = *(const v4f*)(s_stage + i);
      v4bf y;
      y[0] = (__bf16)x[0]; y[1] = (__bf16)x[1]; y[2] = (__bf16)x[2]; y[3] = (__bf16)x[3];
      *(v4bf*)(&s_hs[(i >> 6) * HLD + (i & 63)]) = y;
    }
#else
    __syncthreads();
    {
      const float* h = (e == 0 ? h0 : (e == 1 ? h1 : h2)) + hoff;
      for (int i = tid * 4; i < T_ * HID_; i += NTHREADS * 4) {
        const v4f x = *(const v4f*)(h + i);
        v4bf y;
        y[0] = (__bf16)x[0]; y[1] = (__bf16)x[1]; y[2] = (__bf16)x[2]; y[3] = (__bf16)x[3];
        *(v4bf*)(&s_hs[(i >> 6) * HLD + (i & 63)]) = y;
      }
    }
#endif
    // Weights, transposed: s_wT[mat][m][k] = W[k][m].
#pragma unroll
    for (int mat = 0; mat < 3; ++mat) {
      const float* w = (mat == 0 ? hq : (mat == 1 ? kw : vw)) + (size_t)e * HID_ * MH_;
      for (int i = tid; i < HID_ * MH_; i += NTHREADS)
        s_wT[mat][(i & 31) * WTLD + (i >> 5)] = (__bf16)w[i];
    }
    // Zero K-pad columns of v^T (rows of padded v 48..63).
    for (int i = tid; i < MH_ * 2; i += NTHREADS) {
      const v8bf z = {};
      *(v8bf*)(&s_vT[(i >> 1) * WTLD + T_ + (i & 1) * 8]) = z;
    }
    __syncthreads();

#ifdef HAVE_TDM
    // Staging buffer free: start DMA of the next expert's h tile.
    if (wave == 0 && e + 1 < NEXP_)
      tdm_load_h_tile((e == 0 ? h1 : h2) + hoff, s_stage);
#endif

    // ---- Stage B: q/k/v projections. 18 jobs = {q,k,v} x (3 Mt x 2 Nt), K=64. ----
    for (int job = wave; job < 18; job += 4) {
      const int mat = job / 6, t6 = job % 6;
      const int mt = t6 >> 1, nt = t6 & 1;
      v8f c = {};
#pragma unroll
      for (int ks = 0; ks < 2; ++ks) {
        const v16bf a = frag_a(s_hs, HLD, mt * 16, ks * 32, lane);
        const v16bf b = frag_bT(s_wT[mat], WTLD, ks * 32, nt * 16, lane);
        c = wmma_bf16(a, b, c);
      }
      if (mat == 0)      store_d_bf16_rm(s_q, QLD, mt * 16, nt * 16, lane, c);
      else if (mat == 1) store_d_bf16_rm(s_k, QLD, mt * 16, nt * 16, lane, c);
      else               store_d_bf16_tr(s_vT, WTLD, mt * 16, nt * 16, lane, c);
    }
    __syncthreads();

    // ---- Stage C: energy = q * k^T  [48][48], K = 32. ----
    for (int job = wave; job < 9; job += 4) {
      const int mt = job / 3, st = job % 3;
      const v16bf a = frag_a(s_q, QLD, mt * 16, 0, lane);
      const v16bf b = frag_bT(s_k, QLD, 0, st * 16, lane);
      v8f c = {};
      c = wmma_bf16(a, b, c);
      store_d_f32(s_scr, ELD, mt * 16, st * 16, lane, c);
    }
    __syncthreads();

    // ---- Stage D: row softmax -> bf16 score tile in s_hs (cols 48..63 zero). ----
    for (int t = tid; t < T_; t += NTHREADS) {
      float* row = s_scr + t * ELD;
      float mx = -3.4e38f;
#pragma unroll
      for (int s4 = 0; s4 < T_; s4 += 4) {
        const v4f x = *(const v4f*)(row + s4);
        mx = fmaxf(mx, fmaxf(fmaxf(x[0], x[1]), fmaxf(x[2], x[3])));
      }
      float sum = 0.f;
#pragma unroll
      for (int s = 0; s < T_; ++s) { const float p = __expf(row[s] - mx); row[s] = p; sum += p; }
      const float inv = 1.f / sum;
#pragma unroll
      for (int s4 = 0; s4 < 64; s4 += 4) {
        v4bf y;
        if (s4 < T_) {
          const v4f p = *(const v4f*)(row + s4);
          y[0] = (__bf16)(p[0] * inv); y[1] = (__bf16)(p[1] * inv);
          y[2] = (__bf16)(p[2] * inv); y[3] = (__bf16)(p[3] * inv);
        } else {
          y[0] = y[1] = y[2] = y[3] = (__bf16)0.f;
        }
        *(v4bf*)(&s_hs[t * HLD + s4]) = y;
      }
    }
    __syncthreads();

    // ---- Stage E: att = score * v  [48][32], K padded 48->64. f32 into s_scr. ----
    for (int job = wave; job < 6; job += 4) {
      const int mt = job >> 1, nt = job & 1;
      v8f c = {};
#pragma unroll
      for (int ks = 0; ks < 2; ++ks) {
        const v16bf a = frag_a(s_hs, HLD, mt * 16, ks * 32, lane);
        const v16bf b = frag_bT(s_vT, WTLD, ks * 32, nt * 16, lane);
        c = wmma_bf16(a, b, c);
      }
      store_d_f32(s_scr, ELD, mt * 16, nt * 16, lane, c);
    }
    __syncthreads();

    // ---- Stage F: cosine-similarity routing score -> out[b,n,t,0,e]. ----
    for (int t = tid; t < T_; t += NTHREADS) {
      float num = 0.f, a2 = 0.f;
#pragma unroll
      for (int m4 = 0; m4 < MH_; m4 += 4) {
        const v4f av = *(const v4f*)(s_scr + t * ELD + m4);
#pragma unroll
        for (int j = 0; j < 4; ++j) {
          num += s_memb[t * MLD + m4 + j] * av[j];
          a2  += av[j] * av[j];
        }
      }
      const float sc = num / fmaxf(s_mnorm[t] * sqrtf(a2), EPS_);
      out[((size_t)bn * T_ + t) * NEXP_ + e] = sc;
    }
    __syncthreads();   // buffers reused by next expert
  }
}

// ---------------------------------------------------------------------------
// Host entry
// ---------------------------------------------------------------------------
extern "C" void kernel_launch(void* const* d_in, const int* in_sizes, int n_in,
                              void* d_out, int out_size, void* d_ws, size_t ws_size,
                              hipStream_t stream) {
  (void)in_sizes; (void)n_in; (void)out_size; (void)d_ws; (void)ws_size;
  const float* input  = (const float*)d_in[0];
  const float* h0     = (const float*)d_in[1];
  const float* h1     = (const float*)d_in[2];
  const float* h2     = (const float*)d_in[3];
  const float* memory = (const float*)d_in[4];
  const float* iq     = (const float*)d_in[5];
  const float* hq     = (const float*)d_in[6];
  const float* kw     = (const float*)d_in[7];
  const float* vw     = (const float*)d_in[8];
  float* out = (float*)d_out;

  dim3 grid(B_ * N_);
  dim3 block(NTHREADS);
  hipLaunchKernelGGL(testam_fused, grid, block, 0, stream,
                     input, h0, h1, h2, memory, iq, hq, kw, vw, out);
}
